// GCN_68908455297308
// MI455X (gfx1250) — matrix operand order repaired and checked by reading, not compiled, not run on hardware
//
#include <hip/hip_runtime.h>

#define F 64

typedef __attribute__((ext_vector_type(16))) __bf16 v16bf;
typedef __attribute__((ext_vector_type(8)))  float  v8f;

// ---- column sums (for PairNorm col_mean): hierarchical, few atomics ----
__global__ void colsum_kernel(const float* __restrict__ h, float* __restrict__ colsum,
                              int n, int rows_per_block) {
  __shared__ float part[256];
  int f   = threadIdx.x & 63;
  int sub = threadIdx.x >> 6;             // 0..3
  int start = blockIdx.x * rows_per_block;
  int end   = min(n, start + rows_per_block);
  float s = 0.f;
  for (int i = start + sub; i < end; i += 4)
    s += h[(long)i * F + f];              // 64 consecutive lanes read one row: coalesced
  part[threadIdx.x] = s;
  __syncthreads();
  if (sub == 0)
    unsafeAtomicAdd(colsum + f, part[f] + part[64 + f] + part[128 + f] + part[192 + f]);
}

// ---- per-row inverse norms: one wave32 per row ----
__global__ void rownorm_kernel(const float* __restrict__ h, float* __restrict__ rinv, int n) {
  int wid  = (int)(((long)blockIdx.x * blockDim.x + threadIdx.x) >> 5);
  int lane = threadIdx.x & 31;
  if (wid >= n) return;
  float2 v = ((const float2*)(h + (long)wid * F))[lane];
  float s = v.x * v.x + v.y * v.y;
  s += __shfl_xor(s, 16);
  s += __shfl_xor(s, 8);
  s += __shfl_xor(s, 4);
  s += __shfl_xor(s, 2);
  s += __shfl_xor(s, 1);
  if (lane == 0) rinv[wid] = 1.0f / sqrtf(1e-6f + s);
}

// ---- self-loop + fused PairNorm: agg = h*rinv - mu ----
template <bool NORM>
__global__ void init_agg_kernel(const float* __restrict__ h, const float* __restrict__ rinv,
                                const float* __restrict__ colsum, float invN,
                                float* __restrict__ agg, long total) {
  long idx = (long)blockIdx.x * blockDim.x + threadIdx.x;
  if (idx >= total) return;
  float v = h[idx];
  if (NORM) {
    int  f = (int)(idx & (F - 1));
    long i = idx >> 6;
    v = v * rinv[i] - colsum[f] * invN;
  }
  agg[idx] = v;
}

// ---- edge scatter: 16 lanes per edge, float4 gather + 4 hw f32 atomics ----
template <bool NORM>
__global__ void edge_agg_kernel(const float* __restrict__ h, const float* __restrict__ rinv,
                                const float* __restrict__ colsum, float invN,
                                const int* __restrict__ src, const int* __restrict__ dst,
                                float* __restrict__ agg, int ne) {
  long t = (long)blockIdx.x * blockDim.x + threadIdx.x;
  int  e = (int)(t >> 4);
  if (e >= ne) return;
  int j = (int)(t & 15);
  int s = src[e];
  int d = dst[e];
  float4 hv = ((const float4*)(h + (long)s * F))[j];
  float4 val;
  if (NORM) {
    float  r   = rinv[s];
    float4 mu4 = ((const float4*)colsum)[j];
    val.x = hv.x * r - mu4.x * invN;
    val.y = hv.y * r - mu4.y * invN;
    val.z = hv.z * r - mu4.z * invN;
    val.w = hv.w * r - mu4.w * invN;
  } else {
    val = hv;
  }
  float* ap = agg + (long)d * F + j * 4;
  unsafeAtomicAdd(ap + 0, val.x);
  unsafeAtomicAdd(ap + 1, val.y);
  unsafeAtomicAdd(ap + 2, val.z);
  unsafeAtomicAdd(ap + 3, val.w);
}

// ---- pre-pack W into bf16 hi/lo B-fragments in exact WMMA lane layout ----
// frag[((t*2+ks)*32 + lane)*8 + v] packs W[k..k+1][t*16+lr], k = ks*32 + half*16 + 2v
// hi half stored in frag[0..NFRAG), lo residual in frag[NFRAG..2*NFRAG)
template <int COLS>
__global__ void pack_w_kernel(const float* __restrict__ W, unsigned* __restrict__ frag) {
  const int nfrag = COLS * 32;
  int id = blockIdx.x * blockDim.x + threadIdx.x;
  if (id >= nfrag) return;
  int v    = id & 7;
  int lane = (id >> 3) & 31;
  int ks   = (id >> 8) & 1;
  int t    = id >> 9;
  int half = lane >> 4, lr = lane & 15;
  int k = ks * 32 + half * 16 + 2 * v;
  int n = t * 16 + lr;
  float f0 = W[k * COLS + n], f1 = W[(k + 1) * COLS + n];
  __bf16 h0 = (__bf16)f0, h1 = (__bf16)f1;
  __bf16 l0 = (__bf16)(f0 - (float)h0), l1 = (__bf16)(f1 - (float)h1);
  unsigned hi = ((unsigned)__builtin_bit_cast(unsigned short, h1) << 16) |
                (unsigned)__builtin_bit_cast(unsigned short, h0);
  unsigned lo = ((unsigned)__builtin_bit_cast(unsigned short, l1) << 16) |
                (unsigned)__builtin_bit_cast(unsigned short, l0);
  frag[id]         = hi;
  frag[nfrag + id] = lo;
}

// ---- WMMA GEMM: out[n,COLS] = relu?(A[n,64] @ W[64,COLS] + b) ----
// One wave per 16-row tile. bf16 hi/lo split => ~fp32 accuracy, f32 accumulate.
// B fragments are pre-packed: two 32B vector loads per (tile, kstep) per operand.
template <int COLS, bool RELU>
__global__ void gemm_wmma_kernel(const float* __restrict__ A, const unsigned* __restrict__ wfrag,
                                 const float* __restrict__ bias, float* __restrict__ out,
                                 int nrows) {
  constexpr int NFRAG = COLS * 32;
  int wid  = (int)(((long)blockIdx.x * blockDim.x + threadIdx.x) >> 5);
  int lane = threadIdx.x & 31;
  int row0 = wid * 16;
  if (row0 >= nrows) return;           // whole wave exits: EXEC stays all-ones for WMMA
  int half = lane >> 4;                // 0/1
  int lr   = lane & 15;                // A: row-in-tile, B/D: col-in-tile
  int arow = min(row0 + lr, nrows - 1);
  const float* ap = A + (long)arow * F;

  // A fragments (16x32 bf16 per k-step), contiguous k order: VGPR v holds
  // K = half*16 + v*2 (+1)
  v16bf a_hi[2], a_lo[2];
#pragma unroll
  for (int ks = 0; ks < 2; ++ks) {
#pragma unroll
    for (int v = 0; v < 8; ++v) {
      int k = ks * 32 + half * 16 + 2 * v;
      float f0 = ap[k], f1 = ap[k + 1];
      __bf16 h0 = (__bf16)f0, h1 = (__bf16)f1;
      a_hi[ks][2 * v]     = h0;
      a_hi[ks][2 * v + 1] = h1;
      a_lo[ks][2 * v]     = (__bf16)(f0 - (float)h0);
      a_lo[ks][2 * v + 1] = (__bf16)(f1 - (float)h1);
    }
  }

#pragma unroll
  for (int t = 0; t < COLS / 16; ++t) {
    v8f acc = {};
#pragma unroll
    for (int ks = 0; ks < 2; ++ks) {
      int fb = ((t * 2 + ks) * 32 + lane) * 8;
      v16bf b_hi = *reinterpret_cast<const v16bf*>(wfrag + fb);
      v16bf b_lo = *reinterpret_cast<const v16bf*>(wfrag + NFRAG + fb);
      acc = __builtin_amdgcn_wmma_f32_16x16x32_bf16(false, a_hi[ks], false, b_hi, (short)0, acc, false, false);
      acc = __builtin_amdgcn_wmma_f32_16x16x32_bf16(false, a_lo[ks], false, b_hi, (short)0, acc, false, false);
      acc = __builtin_amdgcn_wmma_f32_16x16x32_bf16(false, a_hi[ks], false, b_lo, (short)0, acc, false, false);
    }
    // D layout: VGPR r holds (M = row0 + r + 8*half, N = t*16 + lr)
    float bv = bias[t * 16 + lr];
#pragma unroll
    for (int r = 0; r < 8; ++r) {
      int m = row0 + r + half * 8;
      if (m < nrows) {
        float x = acc[r] + bv;
        if (RELU) x = fmaxf(x, 0.f);
        out[(long)m * COLS + t * 16 + lr] = x;
      }
    }
  }
}

extern "C" void kernel_launch(void* const* d_in, const int* in_sizes, int n_in,
                              void* d_out, int out_size, void* d_ws, size_t ws_size,
                              hipStream_t stream) {
  const float* features = (const float*)d_in[0];
  const float* W1 = (const float*)d_in[1];
  const float* b1 = (const float*)d_in[2];
  const float* W2 = (const float*)d_in[3];
  const float* b2 = (const float*)d_in[4];
  const float* W3 = (const float*)d_in[5];
  const float* b3 = (const float*)d_in[6];
  const int*  src = (const int*)d_in[7];
  const int*  dst = (const int*)d_in[8];
  int n  = in_sizes[0] / F;
  int ne = in_sizes[7];

  // workspace carve-out (all fits in L2 alongside inputs: ~52 MB)
  char* ws = (char*)d_ws;
  size_t off = 0;
  auto alloc = [&](size_t bytes) -> void* {
    void* p = ws + off;
    off = (off + bytes + 255) & ~(size_t)255;
    return p;
  };
  float*    h_buf  = (float*)alloc((size_t)n * F * sizeof(float));
  float*    agg    = (float*)alloc((size_t)n * F * sizeof(float));
  float*    rinv   = (float*)alloc((size_t)n * sizeof(float));
  float*    colsum = (float*)alloc(F * sizeof(float));
  unsigned* wf1    = (unsigned*)alloc((size_t)64 * 32 * 2 * sizeof(unsigned)); // 16 KB
  unsigned* wf2    = (unsigned*)alloc((size_t)64 * 32 * 2 * sizeof(unsigned));
  unsigned* wf3    = (unsigned*)alloc((size_t)16 * 32 * 2 * sizeof(unsigned)); // 4 KB
  (void)ws_size;

  float invN = 1.0f / (float)n;
  long total = (long)n * F;
  const int blk = 256;
  int grid_elem = (int)((total + blk - 1) / blk);
  int grid_edge = (int)(((long)ne * 16 + blk - 1) / blk);
  int grid_rn   = (int)(((long)n * 32 + blk - 1) / blk);
  int ntiles    = (n + 15) / 16;
  int grid_gemm = (int)(((long)ntiles * 32 + blk - 1) / blk);
  const int rows_per_block = 1024;
  int grid_cs   = (n + rows_per_block - 1) / rows_per_block;

  // one-shot weight fragment packing (bf16 hi/lo, WMMA lane layout)
  pack_w_kernel<64><<<(64 * 32 + blk - 1) / blk, blk, 0, stream>>>(W1, wf1);
  pack_w_kernel<64><<<(64 * 32 + blk - 1) / blk, blk, 0, stream>>>(W2, wf2);
  pack_w_kernel<16><<<(16 * 32 + blk - 1) / blk, blk, 0, stream>>>(W3, wf3);

  const float*    hin   = features;
  const unsigned* wfs[2] = {wf1, wf2};
  const float*    bs[2]  = {b1, b2};
  for (int layer = 0; layer < 2; ++layer) {
    hipMemsetAsync(colsum, 0, F * sizeof(float), stream);
    colsum_kernel<<<grid_cs, blk, 0, stream>>>(hin, colsum, n, rows_per_block);
    rownorm_kernel<<<grid_rn, blk, 0, stream>>>(hin, rinv, n);
    init_agg_kernel<true><<<grid_elem, blk, 0, stream>>>(hin, rinv, colsum, invN, agg, total);
    edge_agg_kernel<true><<<grid_edge, blk, 0, stream>>>(hin, rinv, colsum, invN, src, dst, agg, ne);
    gemm_wmma_kernel<64, true><<<grid_gemm, blk, 0, stream>>>(agg, wfs[layer], bs[layer], h_buf, n);
    hin = h_buf;
  }
  // layer 3: no PairNorm, no ReLU, 16 output cols straight to d_out
  init_agg_kernel<false><<<grid_elem, blk, 0, stream>>>(hin, nullptr, nullptr, 0.f, agg, total);
  edge_agg_kernel<false><<<grid_edge, blk, 0, stream>>>(hin, nullptr, nullptr, 0.f, src, dst, agg, ne);
  gemm_wmma_kernel<16, false><<<grid_gemm, blk, 0, stream>>>(agg, wf3, b3, (float*)d_out, n);
}
